// ASTRM_87668872446293
// MI455X (gfx1250) — matrix operand order repaired and checked by reading, not compiled, not run on hardware
//
#include <hip/hip_runtime.h>
#include <cmath>

// ---------------- problem constants ----------------
#define B_   8
#define T_   8
#define C_   256
#define CH_  64
#define HW_  3136         // 56*56
#define W_   56
#define KSZ  3
#define TH_  16
#define KDIM 768          // 3*C_
#define NTOT (B_*T_*C_*HW_)   // 51,380,224
#define NPIX (B_*T_*HW_)      // 200,704
#define BPITCH4 40            // K4 ldsB column pitch (ushorts), 80B = 16*5
#define BPITCH5 72            // K5 ldsB column pitch (ushorts), 144B = 16*9

// ---------------- vector / fragment plumbing ----------------
typedef __bf16 bf16x16 __attribute__((ext_vector_type(16)));
typedef float  f32x8   __attribute__((ext_vector_type(8)));
typedef float  f32x4   __attribute__((ext_vector_type(4)));
typedef unsigned int u32x4 __attribute__((ext_vector_type(4)));
typedef unsigned int u32x2 __attribute__((ext_vector_type(2)));
union Frag16 { bf16x16 v; unsigned short u[16]; u32x4 q[2]; };

__device__ __forceinline__ unsigned short f2bf(float f) {
  union { float f; unsigned u; } c; c.f = f;
  unsigned r = c.u + 0x7FFFu + ((c.u >> 16) & 1u);   // RNE
  return (unsigned short)(r >> 16);
}
__device__ __forceinline__ float sigmoidf(float x) { return 1.0f / (1.0f + __expf(-x)); }

// ---------------- gfx1250 async global->LDS staging ----------------
#if defined(__AMDGCN__) && __has_builtin(__builtin_amdgcn_global_load_async_to_lds_b128)
#define HAVE_ASYNC_LDS 1
// exact parameter types per hipcc diagnostic: 'int __attribute__((vector_size(16))) __device__ *'
typedef int i32x4v __attribute__((vector_size(16)));
typedef __attribute__((address_space(1))) i32x4v* as1_v4i;   // global (__device__)
typedef __attribute__((address_space(3))) i32x4v* as3_v4i;   // LDS (__shared__)
#endif

__device__ __forceinline__ void wait_async0() {
#if defined(__AMDGCN__)
#if __has_builtin(__builtin_amdgcn_s_wait_asynccnt)
  __builtin_amdgcn_s_wait_asynccnt(0);
#elif defined(HAVE_ASYNC_LDS)
  asm volatile("s_wait_asynccnt 0x0" ::: "memory");
#endif
#endif
}

// regular (unmasked) tile copy: global -> LDS, n_ushort multiple of 8
__device__ __forceinline__ void copy_tile_to_lds(unsigned short* dst, const unsigned short* src,
                                                 int n_ushort, int tid, int nthreads) {
  int chunks = n_ushort >> 3;                       // 16B chunks
#ifdef HAVE_ASYNC_LDS
  for (int j = tid; j < chunks; j += nthreads)
    __builtin_amdgcn_global_load_async_to_lds_b128(
        (as1_v4i)(i32x4v*)(void*)const_cast<unsigned short*>(src + j * 8),
        (as3_v4i)(i32x4v*)(void*)(dst + j * 8), 0, 0);
#else
  for (int j = tid; j < chunks; j += nthreads)
    *(u32x4*)(dst + j * 8) = *(const u32x4*)(src + j * 8);
#endif
}

// ---------------- K0: repack weights to bf16 + fold BN ----------------
__global__ void k0_prep(const float* __restrict__ w_reduce,
                        const float* __restrict__ w_expand,
                        const float* __restrict__ bn_gamma,
                        const float* __restrict__ bn_beta,
                        const float* __restrict__ bn_mean,
                        const float* __restrict__ bn_var,
                        unsigned short* __restrict__ wred_bf,
                        unsigned short* __restrict__ wexp_bf,
                        float* __restrict__ bn_scale,
                        float* __restrict__ bn_bias) {
  int i = blockIdx.x * blockDim.x + threadIdx.x;
  if (i < CH_ * KDIM) {
    int m = i / KDIM, k = i % KDIM;
    int dt = k / C_, ci = k % C_;
    wred_bf[i] = f2bf(w_reduce[(m * C_ + ci) * KSZ + dt]);
  } else if (i - CH_ * KDIM < C_ * CH_) {
    int j = i - CH_ * KDIM;
    wexp_bf[j] = f2bf(w_expand[j]);
  }
  if (i < CH_) {
    float inv = bn_gamma[i] * rsqrtf(bn_var[i] + 1e-5f);
    bn_scale[i] = inv;
    bn_bias[i]  = bn_beta[i] - bn_mean[i] * inv;
  }
}

// ---------------- K1: channel max / mean (b128 vectorized) ----------------
__global__ void k1_pool(const float* __restrict__ x, float* __restrict__ pooled) {
  int i = blockIdx.x * blockDim.x + threadIdx.x;     // one float4 of pixels
  if (i >= NPIX / 4) return;
  int bt = i / (HW_ / 4), p = (i % (HW_ / 4)) * 4;
  const float* base = x + (size_t)bt * C_ * HW_ + p;
  f32x4 mx = {-3.4e38f, -3.4e38f, -3.4e38f, -3.4e38f};
  f32x4 sm = {0.f, 0.f, 0.f, 0.f};
  for (int c = 0; c < C_; ++c) {
    f32x4 v = *(const f32x4*)(base + (size_t)c * HW_);
#pragma unroll
    for (int e = 0; e < 4; ++e) mx[e] = fmaxf(mx[e], v[e]);
    sm += v;
  }
  *(f32x4*)(pooled + (size_t)bt * 2 * HW_ + p)       = mx;
  *(f32x4*)(pooled + (size_t)bt * 2 * HW_ + HW_ + p) = sm * (1.0f / C_);
}

// ---------------- K2: 7x7 conv (2->1) -> gate g = 1 + sigmoid ----------------
__global__ void k2_gate(const float* __restrict__ pooled,
                        const float* __restrict__ w_spatial,
                        float* __restrict__ g) {
  __shared__ float ws[2 * 49];
  if (threadIdx.x < 2 * 49) ws[threadIdx.x] = w_spatial[threadIdx.x];
  __syncthreads();
  int i = blockIdx.x * blockDim.x + threadIdx.x;
  if (i >= NPIX) return;
  int bt = i / HW_, p = i % HW_, y = p / W_, xcol = p % W_;
  const float* pb = pooled + (size_t)bt * 2 * HW_;
  float acc = 0.f;
  for (int ic = 0; ic < 2; ++ic)
    for (int dy = 0; dy < 7; ++dy) {
      int iy = y + dy - 3; if (iy < 0 || iy >= W_) continue;
      for (int dx = 0; dx < 7; ++dx) {
        int ix = xcol + dx - 3; if (ix < 0 || ix >= W_) continue;
        acc += ws[ic * 49 + dy * 7 + dx] * pb[(size_t)ic * HW_ + iy * W_ + ix];
      }
    }
  g[i] = 1.0f + sigmoidf(acc);
}

// ---------------- K3: x_s = x * g (fp32 + bf16 copy), b128 vectorized --------
__global__ void k3_scale(const float* __restrict__ x, const float* __restrict__ g,
                         float* __restrict__ xs, unsigned short* __restrict__ xs_bf) {
  size_t i = (size_t)blockIdx.x * blockDim.x + threadIdx.x;
  if (i >= (size_t)(NTOT / 4)) return;
  size_t idx = i * 4;
  int p  = (int)(idx % HW_);
  int bt = (int)(idx / ((size_t)C_ * HW_));
  f32x4 xv = *(const f32x4*)(x + idx);
  f32x4 gv = *(const f32x4*)(g + (size_t)bt * HW_ + p);
  f32x4 v = xv * gv;
  *(f32x4*)(xs + idx) = v;
  union { unsigned short u[4]; u32x2 d; } o;
#pragma unroll
  for (int e = 0; e < 4; ++e) o.u[e] = f2bf(v[e]);
  *(u32x2*)(xs_bf + idx) = o.d;
}

// ---------------- K4: reduce conv3d as WMMA GEMM (64 x 768 x HW per (b,t)) ----
// grid: (25 n-tiles of 128, 64 (b,t)); block 256 = 8 waves, one 16-col subtile
// per wave, 4 M-tiles per wave. A (64x768 bf16) async-staged in LDS.
// B tiles double-buffered, stored fragment-major (column pitch BPITCH4).
__global__ void k4_reduce_gemm(const unsigned short* __restrict__ xs_bf,
                               const unsigned short* __restrict__ wred_bf,
                               const float* __restrict__ bn_scale,
                               const float* __restrict__ bn_bias,
                               unsigned short* __restrict__ t_bf) {
  extern __shared__ unsigned char smem[];
  unsigned short* ldsA  = (unsigned short*)smem;             // 64*768
  unsigned short* ldsB0 = ldsA + CH_ * KDIM;                 // 128*BPITCH4
  unsigned short* ldsB1 = ldsB0 + 128 * BPITCH4;
  int bt = blockIdx.y, b = bt / T_, t = bt % T_;
  int n0 = blockIdx.x * 128;
  int tid = threadIdx.x, lane = tid & 31, wid = tid >> 5;
  int half = lane >> 4, l16 = lane & 15;

  copy_tile_to_lds(ldsA, wred_bf, CH_ * KDIM, tid, 256);     // async

  // stage one 32x128 B chunk, swizzled to fragment-major (column-major)
  auto stage = [&](int ks, unsigned short* dst) {
    int dt = ks >> 3, ci0 = (ks & 7) * 32, tt = t + dt - 1;
    for (int j = tid; j < 512; j += 256) {                   // 512 x 16B chunks
      int r = j >> 4, c8 = (j & 15) * 8, col = n0 + c8;
      union { unsigned short u[8]; u32x4 q; } vals;
      vals.q = (u32x4){0, 0, 0, 0};
      if (tt >= 0 && tt < T_ && col < HW_)
        vals.q = *(const u32x4*)&xs_bf[(((size_t)(b * T_ + tt) * C_ + (ci0 + r)) * HW_) + col];
#pragma unroll
      for (int e = 0; e < 8; ++e) dst[(c8 + e) * BPITCH4 + r] = vals.u[e];
    }
  };

  stage(0, ldsB0);
  wait_async0();
  __syncthreads();

  f32x8 acc[4] = {};
  for (int ks = 0; ks < 24; ++ks) {
    unsigned short* cur = (ks & 1) ? ldsB1 : ldsB0;
    unsigned short* nxt = (ks & 1) ? ldsB0 : ldsB1;
    if (ks + 1 < 24) stage(ks + 1, nxt);                     // overlap with WMMA

    const unsigned short* bsrc = cur + (wid * 16 + l16) * BPITCH4 + 16 * half;
    Frag16 bf;                                               // B 32x16: K = e + 16*half
    bf.q[0] = *(const u32x4*)bsrc;
    bf.q[1] = *(const u32x4*)(bsrc + 8);
#pragma unroll
    for (int mt = 0; mt < 4; ++mt) {
      const unsigned short* asrc = ldsA + (mt * 16 + l16) * KDIM + ks * 32 + 16 * half;
      Frag16 af;                                             // A 16x32: K = e + 16*half
      af.q[0] = *(const u32x4*)asrc;
      af.q[1] = *(const u32x4*)(asrc + 8);
      acc[mt] = __builtin_amdgcn_wmma_f32_16x16x32_bf16(
          false, af.v, false, bf.v, (short)0, acc[mt], false, false);
    }
    __syncthreads();
  }

  // epilogue: BN + ReLU -> bf16 t  (layout (b, ch, t, p))
  int p = n0 + wid * 16 + l16;
#pragma unroll
  for (int mt = 0; mt < 4; ++mt)
#pragma unroll
    for (int i = 0; i < 8; ++i) {
      int co = mt * 16 + i + 8 * half;                       // C/D: M = i + 8*half
      float val = fmaxf(acc[mt][i] * bn_scale[co] + bn_bias[co], 0.f);
      if (p < HW_)
        t_bf[(((size_t)b * CH_ + co) * T_ + t) * HW_ + p] = f2bf(val);
    }
}

// ---------------- K5: expand GEMM (256 x 64 x HW) + gate + spatial-mean ------
__global__ void k5_expand_gate(const unsigned short* __restrict__ t_bf,
                               const unsigned short* __restrict__ wexp_bf,
                               const float* __restrict__ xs,
                               float* __restrict__ xt,
                               float* __restrict__ vsum) {
  extern __shared__ unsigned char smem[];
  unsigned short* ldsA = (unsigned short*)smem;              // 256*64
  unsigned short* ldsB = ldsA + C_ * CH_;                    // 128*BPITCH5
  int bt = blockIdx.y, b = bt / T_, t = bt % T_;
  int n0 = blockIdx.x * 128;
  int tid = threadIdx.x, lane = tid & 31, wid = tid >> 5;
  int half = lane >> 4, l16 = lane & 15;

  copy_tile_to_lds(ldsA, wexp_bf, C_ * CH_, tid, 256);       // async

  for (int j = tid; j < 1024; j += 256) {                    // 64x128 B tile, swizzled
    int r = j >> 4, c8 = (j & 15) * 8, col = n0 + c8;
    union { unsigned short u[8]; u32x4 q; } vals;
    vals.q = (u32x4){0, 0, 0, 0};
    if (col < HW_)
      vals.q = *(const u32x4*)&t_bf[(((size_t)b * CH_ + r) * T_ + t) * HW_ + col];
#pragma unroll
    for (int e = 0; e < 8; ++e) ldsB[(c8 + e) * BPITCH5 + r] = vals.u[e];
  }
  wait_async0();
  __syncthreads();

  Frag16 bfr[2];
  const unsigned short* bsrc = ldsB + (wid * 16 + l16) * BPITCH5;
#pragma unroll
  for (int kc = 0; kc < 2; ++kc) {
    bfr[kc].q[0] = *(const u32x4*)(bsrc + kc * 32 + 16 * half);
    bfr[kc].q[1] = *(const u32x4*)(bsrc + kc * 32 + 16 * half + 8);
  }

  int p = n0 + wid * 16 + l16;
  for (int mt = 0; mt < 16; ++mt) {
    f32x8 acc = {};
#pragma unroll
    for (int kc = 0; kc < 2; ++kc) {
      const unsigned short* asrc = ldsA + (mt * 16 + l16) * CH_ + kc * 32 + 16 * half;
      Frag16 af;
      af.q[0] = *(const u32x4*)asrc;
      af.q[1] = *(const u32x4*)(asrc + 8);
      acc = __builtin_amdgcn_wmma_f32_16x16x32_bf16(
          false, af.v, false, bfr[kc].v, (short)0, acc, false, false);
    }
#pragma unroll
    for (int i = 0; i < 8; ++i) {
      int c = mt * 16 + i + 8 * half;
      float gate = 1.0f + sigmoidf(acc[i]);
      float xv = (p < HW_) ? xs[((size_t)(b * T_ + t) * C_ + c) * HW_ + p] : 0.f;
      float xo = xv * gate;
      if (p < HW_) xt[(((size_t)b * C_ + c) * T_ + t) * HW_ + p] = xo;
      float s = xo;                                          // 16-lane (half-wave) N-reduction
#pragma unroll
      for (int off = 8; off >= 1; off >>= 1) s += __shfl_xor(s, off, 16);
      if (l16 == 0) atomicAdd(&vsum[((size_t)b * C_ + c) * T_ + t], s);
    }
  }
}

// ---------------- K6: dynamic temporal kernel (tiny FCs + softmax) -----------
__global__ void k6_dyn(const float* __restrict__ vsum,
                       const float* __restrict__ w_fc1, const float* __restrict__ b_fc1,
                       const float* __restrict__ w_fc2, const float* __restrict__ b_fc2,
                       float* __restrict__ kern) {
  int i = blockIdx.x * blockDim.x + threadIdx.x;   // b*C + c
  if (i >= B_ * C_) return;
  float v[T_];
#pragma unroll
  for (int t = 0; t < T_; ++t) v[t] = vsum[(size_t)i * T_ + t] * (1.0f / HW_);
  float h[TH_];
#pragma unroll
  for (int j = 0; j < TH_; ++j) {
    float s = b_fc1[j];
#pragma unroll
    for (int t = 0; t < T_; ++t) s += v[t] * w_fc1[j * T_ + t];
    h[j] = fmaxf(s, 0.f);
  }
  float lg[KSZ], mx = -3.4e38f;
#pragma unroll
  for (int k = 0; k < KSZ; ++k) {
    float s = b_fc2[k];
#pragma unroll
    for (int j = 0; j < TH_; ++j) s += h[j] * w_fc2[k * TH_ + j];
    lg[k] = s; mx = fmaxf(mx, s);
  }
  float den = 0.f;
#pragma unroll
  for (int k = 0; k < KSZ; ++k) { lg[k] = __expf(lg[k] - mx); den += lg[k]; }
#pragma unroll
  for (int k = 0; k < KSZ; ++k) kern[(size_t)i * KSZ + k] = lg[k] / den;
}

// ---------------- K7: depthwise 3-tap temporal stencil (b128 vectorized) -----
__global__ void k7_stencil(const float* __restrict__ xt, const float* __restrict__ kern,
                           float* __restrict__ out) {
  size_t i = (size_t)blockIdx.x * blockDim.x + threadIdx.x;
  if (i >= (size_t)(NTOT / 4)) return;
  size_t idx = i * 4;
  int p  = (int)(idx % HW_);
  int c  = (int)((idx / HW_) % C_);
  int bt = (int)(idx / ((size_t)C_ * HW_));
  int b = bt / T_, t = bt % T_;
  const float* kb = &kern[((size_t)b * C_ + c) * KSZ];
  size_t base = (((size_t)b * C_ + c) * T_ + t) * HW_ + p;
  f32x4 r = kb[1] * *(const f32x4*)(xt + base);
  if (t > 0)      r += kb[0] * *(const f32x4*)(xt + base - HW_);
  if (t < T_ - 1) r += kb[2] * *(const f32x4*)(xt + base + HW_);
  *(f32x4*)(out + idx) = r;
}

// ---------------- host launcher ----------------
extern "C" void kernel_launch(void* const* d_in, const int* in_sizes, int n_in,
                              void* d_out, int out_size, void* d_ws, size_t ws_size,
                              hipStream_t stream) {
  const float* x        = (const float*)d_in[0];
  const float* w_spat   = (const float*)d_in[1];
  const float* w_reduce = (const float*)d_in[2];
  const float* bn_gamma = (const float*)d_in[3];
  const float* bn_beta  = (const float*)d_in[4];
  const float* bn_mean  = (const float*)d_in[5];
  const float* bn_var   = (const float*)d_in[6];
  const float* w_expand = (const float*)d_in[7];
  const float* w_fc1    = (const float*)d_in[8];
  const float* b_fc1    = (const float*)d_in[9];
  const float* w_fc2    = (const float*)d_in[10];
  const float* b_fc2    = (const float*)d_in[11];
  float* out = (float*)d_out;

  // ---- workspace layout ----
  char* ws = (char*)d_ws;
  size_t off = 0;
  auto take = [&](size_t bytes) -> void* {
    void* p = ws + off; off = (off + bytes + 255) & ~(size_t)255; return p;
  };
  float* pooled  = (float*)take((size_t)B_ * T_ * 2 * HW_ * 4);
  float* g       = (float*)take((size_t)NPIX * 4);
  float* xs      = (float*)take((size_t)NTOT * 4);
  float* xt      = (float*)take((size_t)NTOT * 4);
  float* vsum    = (float*)take((size_t)B_ * C_ * T_ * 4);
  float* kern    = (float*)take((size_t)B_ * C_ * KSZ * 4);
  float* bn_scale = (float*)take((size_t)CH_ * 4);
  float* bn_bias  = (float*)take((size_t)CH_ * 4);
  unsigned short* xs_bf   = (unsigned short*)take((size_t)NTOT * 2);
  unsigned short* t_bf    = (unsigned short*)take((size_t)B_ * CH_ * T_ * HW_ * 2);
  unsigned short* wred_bf = (unsigned short*)take((size_t)CH_ * KDIM * 2);
  unsigned short* wexp_bf = (unsigned short*)take((size_t)C_ * CH_ * 2);

  const size_t smem_k4 = (size_t)CH_ * KDIM * 2 + 2 * 128 * BPITCH4 * 2;  // ~116 KB
  const size_t smem_k5 = (size_t)C_ * CH_ * 2 + 128 * BPITCH5 * 2;        // ~50 KB
  (void)hipFuncSetAttribute((const void*)k4_reduce_gemm,
      hipFuncAttributeMaxDynamicSharedMemorySize, (int)smem_k4);
  (void)hipFuncSetAttribute((const void*)k5_expand_gate,
      hipFuncAttributeMaxDynamicSharedMemorySize, (int)smem_k5);

  k0_prep<<<(CH_ * KDIM + C_ * CH_) / 256, 256, 0, stream>>>(
      w_reduce, w_expand, bn_gamma, bn_beta, bn_mean, bn_var,
      wred_bf, wexp_bf, bn_scale, bn_bias);
  k1_pool<<<NPIX / 4 / 256, 256, 0, stream>>>(x, pooled);
  k2_gate<<<NPIX / 256, 256, 0, stream>>>(pooled, w_spat, g);
  k3_scale<<<NTOT / 4 / 256, 256, 0, stream>>>(x, g, xs, xs_bf);
  (void)hipMemsetAsync(vsum, 0, (size_t)B_ * C_ * T_ * 4, stream);
  k4_reduce_gemm<<<dim3(25, B_ * T_), 256, smem_k4, stream>>>(
      xs_bf, wred_bf, bn_scale, bn_bias, t_bf);
  k5_expand_gate<<<dim3(25, B_ * T_), 256, smem_k5, stream>>>(t_bf, wexp_bf, xs, xt, vsum);
  k6_dyn<<<(B_ * C_ + 255) / 256, 256, 0, stream>>>(vsum, w_fc1, b_fc1, w_fc2, b_fc2, kern);
  k7_stencil<<<NTOT / 4 / 256, 256, 0, stream>>>(xt, kern, out);
}
